// GRUDModel_82575041232970
// MI455X (gfx1250) — matrix operand order repaired and checked by reading, not compiled, not run on hardware
//
#include <hip/hip_runtime.h>
#include <hip/hip_bf16.h>
#include <math.h>

namespace {
constexpr int kN    = 10000;
constexpr int kB    = 16;
constexpr int kT    = 12;
constexpr int kAUX  = 11;
constexpr int kMOE  = 8;
constexpr int kHID  = 64;
constexpr int kDF   = 32;
constexpr int kHD   = 8;
constexpr int kE    = 160000;
constexpr int kBN   = kB * kN;      // 160000
constexpr int kG3   = 3 * kHID;     // 192
constexpr int kTiles = kG3 / 16;    // 12 output tiles of 16 cols
}

typedef __attribute__((ext_vector_type(16))) __bf16 v16bf;
typedef __attribute__((ext_vector_type(8)))  float  v8f;

// ---------- scalar helpers ----------
__device__ __forceinline__ unsigned short f2bf(float f) {
  unsigned u = __float_as_uint(f);
  u += 0x7fffu + ((u >> 16) & 1u);          // round-to-nearest-even
  return (unsigned short)(u >> 16);
}
__device__ __forceinline__ float bf2f(unsigned short u) {
  return __uint_as_float(((unsigned)u) << 16);
}
__device__ __forceinline__ __bf16 us2bf(unsigned short u) {
  union { unsigned short s; __bf16 b; } c; c.s = u; return c.b;
}
// monotonic float <-> uint encoding for atomic segment-max (init 0 < any real)
__device__ __forceinline__ unsigned fenc(float f) {
  unsigned u = __float_as_uint(f);
  return (u & 0x80000000u) ? ~u : (u | 0x80000000u);
}
__device__ __forceinline__ float fdec(unsigned e) {
  unsigned u = (e & 0x80000000u) ? (e & 0x7fffffffu) : ~e;
  return __uint_as_float(u);
}
__device__ __forceinline__ float sigf(float x) { return 1.f / (1.f + __expf(-x)); }

// ---------- 0. pack W_hh [192,64] f32 -> bf16 B-fragments (B 32x16, contiguous k = 16*h + s per chunk) ----------
// flat = (((n0*2 + c)*32 + lane)*8 + i)*2 + p ; element = Whh[o=n0*16+(lane&15)][k=c*32+(lane>>4)*16+2i+p]
__global__ void k_prep_whh(const float* __restrict__ whh, unsigned short* __restrict__ frag) {
  int idx = blockIdx.x * blockDim.x + threadIdx.x;
  if (idx >= kTiles * 2 * 32 * 16) return;
  int p    = idx & 1;
  int i    = (idx >> 1) & 7;
  int lane = (idx >> 4) & 31;
  int c    = (idx >> 9) & 1;
  int n0   = idx >> 10;
  int o = n0 * 16 + (lane & 15);
  int k = c * 32 + (lane >> 4) * 16 + 2 * i + p;
  frag[idx] = f2bf(whh[o * kHID + k]);
}

// ---------- 1. align linears: f_s [N,32] from concat(Fp,Fa), f_p from Fp, f_a from Fa ----------
__global__ void k_align(const float* __restrict__ Fp, const float* __restrict__ Fa,
                        const float* __restrict__ wsw, const float* __restrict__ wsb,
                        const float* __restrict__ wpw, const float* __restrict__ wpb,
                        const float* __restrict__ waw, const float* __restrict__ wab,
                        float* __restrict__ fs, float* __restrict__ fp, float* __restrict__ fa) {
  int n = blockIdx.x * blockDim.x + threadIdx.x;
  if (n >= kN) return;
  float xc[14];
#pragma unroll
  for (int k = 0; k < 3; ++k) xc[k] = Fp[n * 3 + k];
#pragma unroll
  for (int k = 0; k < 11; ++k) xc[3 + k] = Fa[n * 11 + k];
  for (int i = 0; i < kDF; ++i) {
    float as = wsb[i], ap = wpb[i], aa = wab[i];
#pragma unroll
    for (int k = 0; k < 14; ++k) as += wsw[i * 14 + k] * xc[k];
#pragma unroll
    for (int k = 0; k < 3;  ++k) ap += wpw[i * 3 + k] * xc[k];
#pragma unroll
    for (int k = 0; k < 11; ++k) aa += waw[i * 11 + k] * xc[3 + k];
    fs[n * kDF + i] = as; fp[n * kDF + i] = ap; fa[n * kDF + i] = aa;
  }
}

// ---------- 2a. GAT prep: xw = f@W.T, per-head src/dst scores, init accumulators ----------
__global__ void k_gat_prep(const float* __restrict__ f, const float* __restrict__ w,
                           const float* __restrict__ asrc, const float* __restrict__ adst,
                           float* __restrict__ xw, float* __restrict__ ssrc, float* __restrict__ sdst,
                           unsigned* __restrict__ mxe, float* __restrict__ den, float* __restrict__ num) {
  int n = blockIdx.x * blockDim.x + threadIdx.x;
  if (n >= kN) return;
  float xv[kDF], ov[kDF];
#pragma unroll
  for (int k = 0; k < kDF; ++k) xv[k] = f[n * kDF + k];
  for (int i = 0; i < kDF; ++i) {
    float a = 0.f;
#pragma unroll
    for (int k = 0; k < kDF; ++k) a += w[i * kDF + k] * xv[k];
    ov[i] = a;
    xw[n * kDF + i] = a;
    num[n * kDF + i] = 0.f;
  }
  for (int h = 0; h < 4; ++h) {
    float ss = 0.f, sd = 0.f;
#pragma unroll
    for (int d = 0; d < kHD; ++d) {
      ss += ov[h * kHD + d] * asrc[h * kHD + d];
      sd += ov[h * kHD + d] * adst[h * kHD + d];
    }
    ssrc[n * 4 + h] = ss; sdst[n * 4 + h] = sd;
    mxe[n * 4 + h] = 0u; den[n * 4 + h] = 0.f;
  }
}

// ---------- 2b. edge pass 1: segment max via encoded atomicMax ----------
__global__ void k_gat_e1(const int* __restrict__ ei, const float* __restrict__ ssrc,
                         const float* __restrict__ sdst, unsigned* __restrict__ mxe) {
  int m = blockIdx.x * blockDim.x + threadIdx.x;
  int M = kE + kN;
  if (m >= M) return;
  int src = (m < kE) ? ei[m] : (m - kE);
  int dst = (m < kE) ? ei[kE + m] : (m - kE);
#pragma unroll
  for (int h = 0; h < 4; ++h) {
    float e = ssrc[src * 4 + h] + sdst[dst * 4 + h];
    e = (e > 0.f) ? e : 0.2f * e;                       // leaky_relu 0.2
    atomicMax(&mxe[dst * 4 + h], fenc(e));
  }
}

// ---------- 2c. edge pass 2: den += ee ; num += ee * xw[src] ----------
__global__ void k_gat_e2(const int* __restrict__ ei, const float* __restrict__ ssrc,
                         const float* __restrict__ sdst, const float* __restrict__ xw,
                         const unsigned* __restrict__ mxe, float* __restrict__ den,
                         float* __restrict__ num) {
  int m = blockIdx.x * blockDim.x + threadIdx.x;
  int M = kE + kN;
  if (m >= M) return;
  int src = (m < kE) ? ei[m] : (m - kE);
  int dst = (m < kE) ? ei[kE + m] : (m - kE);
#pragma unroll
  for (int h = 0; h < 4; ++h) {
    float e = ssrc[src * 4 + h] + sdst[dst * 4 + h];
    e = (e > 0.f) ? e : 0.2f * e;
    float ee = expf(e - fdec(mxe[dst * 4 + h]));
    atomicAdd(&den[dst * 4 + h], ee);
#pragma unroll
    for (int d = 0; d < kHD; ++d)
      atomicAdd(&num[dst * kDF + h * kHD + d], ee * xw[src * kDF + h * kHD + d]);
  }
}

// ---------- 2d. finalize: out = elu(num/den + bias) ----------
__global__ void k_gat_fin(const float* __restrict__ num, const float* __restrict__ den,
                          const float* __restrict__ bias, float* __restrict__ hout) {
  int n = blockIdx.x * blockDim.x + threadIdx.x;
  if (n >= kN) return;
  for (int h = 0; h < 4; ++h) {
    float dv = den[n * 4 + h] + 1e-16f;
#pragma unroll
    for (int d = 0; d < kHD; ++d) {
      int i = h * kHD + d;
      float v = num[n * kDF + i] / dv + bias[i];
      hout[n * kDF + i] = (v > 0.f) ? v : expm1f(v);    // elu
    }
  }
}

// ---------- 3. fuse MLP -> h0 [N,64] ----------
__global__ void k_fuse(const float* __restrict__ hs, const float* __restrict__ hp,
                       const float* __restrict__ ha,
                       const float* __restrict__ f1w, const float* __restrict__ f1b,
                       const float* __restrict__ f2w, const float* __restrict__ f2b,
                       const float* __restrict__ h01w, const float* __restrict__ h01b,
                       const float* __restrict__ h02w, const float* __restrict__ h02b,
                       float* __restrict__ h0) {
  int n = blockIdx.x * blockDim.x + threadIdx.x;
  if (n >= kN) return;
  float c[96], t1[32], hg[32], t2[32];
#pragma unroll
  for (int k = 0; k < 32; ++k) { c[k] = hs[n*32+k]; c[32+k] = hp[n*32+k]; c[64+k] = ha[n*32+k]; }
  for (int i = 0; i < 32; ++i) {
    float a = f1b[i];
    for (int k = 0; k < 96; ++k) a += f1w[i * 96 + k] * c[k];
    t1[i] = fmaxf(a, 0.f);
  }
  for (int i = 0; i < 32; ++i) {
    float a = f2b[i];
#pragma unroll
    for (int k = 0; k < 32; ++k) a += f2w[i * 32 + k] * t1[k];
    hg[i] = a;
  }
  for (int i = 0; i < 32; ++i) {
    float a = h01b[i];
#pragma unroll
    for (int k = 0; k < 32; ++k) a += h01w[i * 32 + k] * hg[k];
    t2[i] = fmaxf(a, 0.f);
  }
  for (int i = 0; i < kHID; ++i) {
    float a = h02b[i];
#pragma unroll
    for (int k = 0; k < 32; ++k) a += h02w[i * 32 + k] * t2[k];
    h0[n * kHID + i] = a;
  }
}

// ---------- 4. broadcast h0 -> bf16 h[BN,64]; zero decay state ----------
__global__ void k_bcast(const float* __restrict__ h0, unsigned short* __restrict__ hbuf,
                        float* __restrict__ xlast, float* __restrict__ tau) {
  int idx = blockIdx.x * blockDim.x + threadIdx.x;
  if (idx >= kBN * kHID) return;
  int row = idx >> 6, j = idx & 63;
  int n = row % kN;
  hbuf[idx] = f2bf(h0[n * kHID + j]);
  if (j == 0) { xlast[row] = 0.f; tau[row] = 0.f; }
}

// ---------- 5. fused GRU-D step: MoE + decay + WMMA (h @ Whh^T) + gate update ----------
__global__ void __launch_bounds__(32)
k_gru_step(int t,
           const float* __restrict__ X, const float* __restrict__ Y, const float* __restrict__ Msk,
           const float* __restrict__ gate_w, const float* __restrict__ gate_b,
           const float* __restrict__ ew1, const float* __restrict__ eb1,
           const float* __restrict__ ew2, const float* __restrict__ eb2,
           const float* __restrict__ Wd, const float* __restrict__ bd,
           const float* __restrict__ wih, const float* __restrict__ bih,
           const float* __restrict__ bhh,
           const unsigned short* __restrict__ whhF,
           unsigned short* __restrict__ hbuf,
           float* __restrict__ xlast, float* __restrict__ tau) {
  __shared__ float sXt[16][9];
  __shared__ float sZ[2][16][kMOE];
  __shared__ float sGh[16][kG3];

  const int lane = threadIdx.x;
  const int rowBase = blockIdx.x * 16;

  // ---- phase 1: MoE, 2 experts per lane-half, 16 rows per wave ----
  {
    int r = lane & 15, eh = lane >> 4;
    int row = rowBase + r;
    int b = row / kN, n = row % kN;
    const float* x = X + ((size_t)(b * kT + t) * kN + n) * kAUX;
    float xv[kAUX];
#pragma unroll
    for (int k = 0; k < kAUX; ++k) xv[k] = x[k];
    float lg[4];
#pragma unroll
    for (int e = 0; e < 4; ++e) {
      float a = gate_b[e];
#pragma unroll
      for (int k = 0; k < kAUX; ++k) a += gate_w[e * kAUX + k] * xv[k];
      lg[e] = a;
    }
    float mx = fmaxf(fmaxf(lg[0], lg[1]), fmaxf(lg[2], lg[3]));
    float ge[4], s = 0.f;
#pragma unroll
    for (int e = 0; e < 4; ++e) { ge[e] = __expf(lg[e] - mx); s += ge[e]; }
    float inv = 1.f / s;
    float zacc[kMOE];
#pragma unroll
    for (int o = 0; o < kMOE; ++o) zacc[o] = 0.f;
    for (int e = eh * 2; e < eh * 2 + 2; ++e) {
      float g = ge[e] * inv;
      float outv[kMOE];
#pragma unroll
      for (int o = 0; o < kMOE; ++o) outv[o] = eb2[e * kMOE + o];
      for (int h = 0; h < 32; ++h) {
        float hm = eb1[e * 32 + h];
#pragma unroll
        for (int k = 0; k < kAUX; ++k) hm += ew1[(e * 32 + h) * kAUX + k] * xv[k];
        hm = fmaxf(hm, 0.f);
#pragma unroll
        for (int o = 0; o < kMOE; ++o) outv[o] += ew2[(e * kMOE + o) * 32 + h] * hm;
      }
#pragma unroll
      for (int o = 0; o < kMOE; ++o) zacc[o] += g * outv[o];
    }
#pragma unroll
    for (int o = 0; o < kMOE; ++o) sZ[eh][r][o] = zacc[o];
  }
  __syncthreads();

  // ---- phase 2: GRU-D decay (only feature 0 carries state; mask==1 features pass through) ----
  if (lane < 16) {
    int r = lane;
    int row = rowBase + r;
    int b = row / kN, n = row % kN;
    size_t yi = (size_t)(b * kT + t) * kN + n;
    float ph = Y[yi], m1 = Msk[yi];
    float ct = (float)(t + 1);
    float tl = tau[row], xl = xlast[row];
    float spW0 = log1pf(expf(Wd[0]));
    float spb0 = log1pf(expf(bd[0]));
    float dlt = log1pf(fminf(fmaxf(ct - tl + 1e-6f, 0.f), 32.f));
    float gm = __expf(-(spW0 * dlt + spb0));
    float xt0;
    if (m1 > 0.f) { xl = ph; tl = ct; xt0 = ph; }
    else          { xt0 = gm * ph + (1.f - gm) * xl; }
    tau[row] = tl; xlast[row] = xl;
    sXt[r][0] = xt0;
#pragma unroll
    for (int o = 0; o < kMOE; ++o) sXt[r][1 + o] = sZ[0][r][o] + sZ[1][r][o];
  }
  __syncthreads();

  // ---- phase 3: gh = h @ Whh^T via v_wmma_f32_16x16x32_bf16 (2 K-chunks x 12 N-tiles) ----
  {
    v16bf a0, a1;
    const int m = lane & 15, kh = lane >> 4;
    const unsigned short* hrow = hbuf + (size_t)(rowBase + m) * kHID;
#pragma unroll
    for (int i = 0; i < 8; ++i) {
      int koff = kh * 16 + i * 2;   // 16-bit A 16x32, contiguous k = 16*kh + s (matches B pack)
      unsigned w0 = *(const unsigned*)(hrow + koff);
      unsigned w1 = *(const unsigned*)(hrow + 32 + koff);
      a0[2 * i] = us2bf((unsigned short)(w0 & 0xffffu));
      a0[2 * i + 1] = us2bf((unsigned short)(w0 >> 16));
      a1[2 * i] = us2bf((unsigned short)(w1 & 0xffffu));
      a1[2 * i + 1] = us2bf((unsigned short)(w1 >> 16));
    }
    __builtin_prefetch(whhF + (size_t)lane * 16, 0, 1);       // global_prefetch path
    for (int n0 = 0; n0 < kTiles; ++n0) {
      v16bf b0 = *(const v16bf*)(whhF + ((size_t)(n0 * 2 + 0) * 32 + lane) * 16);
      v16bf b1 = *(const v16bf*)(whhF + ((size_t)(n0 * 2 + 1) * 32 + lane) * 16);
      v8f acc = {};
      acc = __builtin_amdgcn_wmma_f32_16x16x32_bf16(false, a0, false, b0, (short)0, acc, false, false);
      acc = __builtin_amdgcn_wmma_f32_16x16x32_bf16(false, a1, false, b1, (short)0, acc, false, false);
      int ncol = n0 * 16 + (lane & 15);
      int mb = (lane >> 4) * 8;
#pragma unroll
      for (int i = 0; i < 8; ++i) sGh[mb + i][ncol] = acc[i]; // f32 C/D 16x16 layout
    }
  }
  __syncthreads();

  // ---- phase 4: gi (9-wide dots) + gates + bf16 h update ----
  for (int it = 0; it < 32; ++it) {
    int idx = it * 32 + lane;
    int m = idx >> 6, j = idx & 63;
    int row = rowBase + m;
    float gr = bih[j], gz = bih[64 + j], gn = bih[128 + j];
#pragma unroll
    for (int k = 0; k < 9; ++k) {
      float xv = sXt[m][k];
      gr += wih[j * 9 + k] * xv;
      gz += wih[(64 + j) * 9 + k] * xv;
      gn += wih[(128 + j) * 9 + k] * xv;
    }
    float hr = sGh[m][j] + bhh[j];
    float hz = sGh[m][64 + j] + bhh[64 + j];
    float hn = sGh[m][128 + j] + bhh[128 + j];
    float rg = sigf(gr + hr);
    float zg = sigf(gz + hz);
    float ng = tanhf(gn + rg * hn);
    float hold = bf2f(hbuf[(size_t)row * kHID + j]);
    float hnew = (1.f - zg) * ng + zg * hold;
    hbuf[(size_t)row * kHID + j] = f2bf(hnew);
  }
}

// ---------- 6. prediction head ----------
__global__ void k_pred(const unsigned short* __restrict__ hbuf, const float* __restrict__ pw,
                       const float* __restrict__ pb, float* __restrict__ out) {
  int row = blockIdx.x * blockDim.x + threadIdx.x;
  if (row >= kBN) return;
  float a = pb[0];
  const unsigned short* hr = hbuf + (size_t)row * kHID;
#pragma unroll
  for (int j = 0; j < kHID; ++j) a += bf2f(hr[j]) * pw[j];
  out[row] = a;
}

extern "C" void kernel_launch(void* const* d_in, const int* in_sizes, int n_in,
                              void* d_out, int out_size, void* d_ws, size_t ws_size,
                              hipStream_t stream) {
  (void)in_sizes; (void)n_in; (void)out_size; (void)ws_size;
  const float* X   = (const float*)d_in[0];
  const float* Y   = (const float*)d_in[1];
  const float* Msk = (const float*)d_in[2];
  const float* Fp  = (const float*)d_in[3];
  const float* Fa  = (const float*)d_in[4];
  const int* es = (const int*)d_in[5];
  const int* ep = (const int*)d_in[6];
  const int* ea = (const int*)d_in[7];
  const float* align_s_w = (const float*)d_in[8];
  const float* align_s_b = (const float*)d_in[9];
  const float* align_p_w = (const float*)d_in[10];
  const float* align_p_b = (const float*)d_in[11];
  const float* align_a_w = (const float*)d_in[12];
  const float* align_a_b = (const float*)d_in[13];
  const float* fuse1_w = (const float*)d_in[14];
  const float* fuse1_b = (const float*)d_in[15];
  const float* fuse2_w = (const float*)d_in[16];
  const float* fuse2_b = (const float*)d_in[17];
  const float* h01_w = (const float*)d_in[18];
  const float* h01_b = (const float*)d_in[19];
  const float* h02_w = (const float*)d_in[20];
  const float* h02_b = (const float*)d_in[21];
  const float* gate_w = (const float*)d_in[22];
  const float* gate_b = (const float*)d_in[23];
  const float* ew1 = (const float*)d_in[24];
  const float* eb1 = (const float*)d_in[25];
  const float* ew2 = (const float*)d_in[26];
  const float* eb2 = (const float*)d_in[27];
  const float* W_decay = (const float*)d_in[28];
  const float* b_decay = (const float*)d_in[29];
  const float* gru_w_ih = (const float*)d_in[30];
  const float* gru_w_hh = (const float*)d_in[31];
  const float* gru_b_ih = (const float*)d_in[32];
  const float* gru_b_hh = (const float*)d_in[33];
  const float* pred_w = (const float*)d_in[34];
  const float* pred_b = (const float*)d_in[35];
  const float* gat_w[3]    = {(const float*)d_in[36], (const float*)d_in[40], (const float*)d_in[44]};
  const float* gat_asrc[3] = {(const float*)d_in[37], (const float*)d_in[41], (const float*)d_in[45]};
  const float* gat_adst[3] = {(const float*)d_in[38], (const float*)d_in[42], (const float*)d_in[46]};
  const float* gat_b[3]    = {(const float*)d_in[39], (const float*)d_in[43], (const float*)d_in[47]};
  const int* eiArr[3] = {es, ep, ea};

  char* ws = (char*)d_ws;
  size_t off = 0;
  auto take = [&](size_t bytes) -> char* {
    char* p = ws + off;
    off = (off + bytes + 255) & ~(size_t)255;
    return p;
  };
  unsigned short* whhF = (unsigned short*)take((size_t)kHID * kG3 * sizeof(unsigned short));
  float* fS = (float*)take((size_t)kN * kDF * 4);
  float* fP = (float*)take((size_t)kN * kDF * 4);
  float* fA = (float*)take((size_t)kN * kDF * 4);
  float* hS = (float*)take((size_t)kN * kDF * 4);
  float* hP = (float*)take((size_t)kN * kDF * 4);
  float* hA = (float*)take((size_t)kN * kDF * 4);
  float* xw = (float*)take((size_t)kN * kDF * 4);
  float* ssrc = (float*)take((size_t)kN * 4 * 4);
  float* sdst = (float*)take((size_t)kN * 4 * 4);
  unsigned* mxe = (unsigned*)take((size_t)kN * 4 * 4);
  float* den = (float*)take((size_t)kN * 4 * 4);
  float* num = (float*)take((size_t)kN * kDF * 4);
  float* h0  = (float*)take((size_t)kN * kHID * 4);
  unsigned short* hbuf = (unsigned short*)take((size_t)kBN * kHID * sizeof(unsigned short));
  float* xlast = (float*)take((size_t)kBN * 4);
  float* taub  = (float*)take((size_t)kBN * 4);

  const float* fIn[3] = {fS, fP, fA};
  float* hOut[3] = {hS, hP, hA};
  const int M = kE + kN;

  k_prep_whh<<<(kHID * kG3 + 255) / 256, 256, 0, stream>>>(gru_w_hh, whhF);
  k_align<<<(kN + 255) / 256, 256, 0, stream>>>(Fp, Fa, align_s_w, align_s_b,
                                                align_p_w, align_p_b, align_a_w, align_a_b,
                                                fS, fP, fA);
  for (int g = 0; g < 3; ++g) {
    k_gat_prep<<<(kN + 255) / 256, 256, 0, stream>>>(fIn[g], gat_w[g], gat_asrc[g], gat_adst[g],
                                                     xw, ssrc, sdst, mxe, den, num);
    k_gat_e1<<<(M + 255) / 256, 256, 0, stream>>>(eiArr[g], ssrc, sdst, mxe);
    k_gat_e2<<<(M + 255) / 256, 256, 0, stream>>>(eiArr[g], ssrc, sdst, xw, mxe, den, num);
    k_gat_fin<<<(kN + 255) / 256, 256, 0, stream>>>(num, den, gat_b[g], hOut[g]);
  }
  k_fuse<<<(kN + 255) / 256, 256, 0, stream>>>(hS, hP, hA, fuse1_w, fuse1_b, fuse2_w, fuse2_b,
                                               h01_w, h01_b, h02_w, h02_b, h0);
  k_bcast<<<(kBN * kHID + 255) / 256, 256, 0, stream>>>(h0, hbuf, xlast, taub);
  for (int t = 0; t < kT; ++t) {
    k_gru_step<<<kBN / 16, 32, 0, stream>>>(t, X, Y, Msk, gate_w, gate_b, ew1, eb1, ew2, eb2,
                                            W_decay, b_decay, gru_w_ih, gru_b_ih, gru_b_hh,
                                            whhF, hbuf, xlast, taub);
  }
  k_pred<<<(kBN + 255) / 256, 256, 0, stream>>>(hbuf, pred_w, pred_b, (float*)d_out);
}
